// IK_NET_70033736728764
// MI455X (gfx1250) — matrix operand order repaired and checked by reading, not compiled, not run on hardware
//
#include <hip/hip_runtime.h>

#define TM      128
#define NTHR    256
#define TDIM    4096
#define NB      64
#define DECC    176
#define OFFC    66
#define SPC     48
#define HSTRIDE 130     // bf16 elems, 260B = 65 dwords (odd) -> conflict-friendly
#define YSTRIDE 33      // f32 elems

typedef __bf16 bf16;
typedef bf16  v16bf __attribute__((ext_vector_type(16)));
typedef float v8f   __attribute__((ext_vector_type(8)));
typedef float v4f   __attribute__((ext_vector_type(4)));

// ---------------- static index tables (expanded from the Python reference) ---
// src: 0=offsets(66ch), 1=dec(176ch), 2=sparse(48ch). Limb sparse +8 pre-baked.
struct Seg  { unsigned char src, cnt; short start; };
struct OSeg { short start, cnt; };

__constant__ int c_nseg[5] = {13, 6, 6, 9, 9};
__constant__ Seg c_segs[5][13] = {
  // main: off 27..65 | dec 0-7,72-103,112-135,144-167,152-159,144-151,88-95,80-87,72-79,0-7 | sp 0-7,24-47
  { {0,39,27},{1,8,0},{1,32,72},{1,24,112},{1,24,144},{1,8,152},{1,8,144},{1,8,88},{1,8,80},{1,8,72},{1,8,0},{2,8,0},{2,24,24} },
  // lleg
  { {0,9,6},{1,8,24},{1,8,16},{1,8,8},{1,8,0},{2,8,8},{0,0,0},{0,0,0},{0,0,0},{0,0,0},{0,0,0},{0,0,0},{0,0,0} },
  // rleg
  { {0,9,18},{1,8,56},{1,8,48},{1,8,40},{1,8,0},{2,8,16},{0,0,0},{0,0,0},{0,0,0},{0,0,0},{0,0,0},{0,0,0},{0,0,0} },
  // larm
  { {0,9,45},{1,8,128},{1,8,120},{1,8,112},{1,8,88},{1,8,80},{1,8,72},{1,8,0},{2,8,32},{0,0,0},{0,0,0},{0,0,0},{0,0,0} },
  // rarm
  { {0,9,57},{1,8,160},{1,8,152},{1,8,144},{1,8,88},{1,8,80},{1,8,72},{1,8,0},{2,8,40},{0,0,0},{0,0,0},{0,0,0},{0,0,0} },
};
__constant__ short c_cin[5]  = {207, 49, 49, 73, 73};
__constant__ short c_kpad[5] = {224, 64, 64, 96, 96};
__constant__ short c_kst[5]  = {226, 66, 66, 98, 98};   // LDS row stride (bf16 elems), 2 mod 4
__constant__ short c_cout[5][3] = {{8,24,24},{8,16,24},{8,16,24},{8,16,24},{8,16,24}};
__constant__ short c_cpad[5][3] = {{16,32,32},{16,16,32},{16,16,32},{16,16,32},{16,16,32}};
__constant__ OSeg c_oseg[5][3][2] = {
  { {{88,8},{0,0}},  {{80,24},{0,0}},  {{0,8},{72,16}} },   // main
  { {{8,8},{0,0}},   {{8,16},{0,0}},   {{8,24},{0,0}}  },   // lleg
  { {{40,8},{0,0}},  {{40,16},{0,0}},  {{40,24},{0,0}} },   // rleg
  { {{112,8},{0,0}}, {{112,16},{0,0}}, {{112,24},{0,0}} },  // larm
  { {{144,8},{0,0}}, {{144,16},{0,0}}, {{144,24},{0,0}} },  // rarm
};

// ------------- workspace layout: per (net,step): W1p | W2p | W3p | b3p -------
__host__ __device__ inline void ws_offsets(int net, int step, size_t& w1, size_t& w2,
                                           size_t& w3, size_t& b3) {
  const int kpadA[5] = {224,64,64,96,96};
  const int cpadA[5][3] = {{16,32,32},{16,16,32},{16,16,32},{16,16,32},{16,16,32}};
  size_t off = 0;
  for (int n = 0; n < 5; ++n)
    for (int s = 0; s < 3; ++s) {
      size_t s1 = (size_t)128 * kpadA[n] * 2;
      size_t s2 = (size_t)128 * 128 * 2;
      size_t s3 = (size_t)cpadA[n][s] * 128 * 2;
      size_t s4 = (size_t)cpadA[n][s] * 4;
      if (n == net && s == step) { w1 = off; w2 = off + s1; w3 = off + s1 + s2; b3 = off + s1 + s2 + s3; return; }
      off += s1 + s2 + s3 + s4;
    }
  w1 = w2 = w3 = b3 = 0;
}

__device__ inline unsigned short f2bf(float x) {   // round-to-nearest-even
  unsigned u = __float_as_uint(x);
  unsigned r = u + 0x7FFFu + ((u >> 16) & 1u);
  return (unsigned short)(r >> 16);
}

// ---------------- weight packing: fp32 -> bf16, K padded, cout padded --------
__global__ void pack_weights_kernel(const float* __restrict__ w1, const float* __restrict__ w2,
                                    const float* __restrict__ w3, const float* __restrict__ b3,
                                    unsigned char* __restrict__ ws, int net, int step) {
  const int cin = c_cin[net], kpad = c_kpad[net];
  const int cout = c_cout[net][step], cpad = c_cpad[net][step];
  size_t o1, o2, o3, ob; ws_offsets(net, step, o1, o2, o3, ob);
  unsigned short* W1 = (unsigned short*)(ws + o1);
  unsigned short* W2 = (unsigned short*)(ws + o2);
  unsigned short* W3 = (unsigned short*)(ws + o3);
  float* B3 = (float*)(ws + ob);
  const int tid = blockIdx.x * blockDim.x + threadIdx.x;
  const int stride = gridDim.x * blockDim.x;
  for (int i = tid; i < 128 * kpad; i += stride) {
    int r = i / kpad, c = i - r * kpad;
    W1[i] = f2bf(c < cin ? w1[r * cin + c] : 0.f);
  }
  for (int i = tid; i < 128 * 128; i += stride) W2[i] = f2bf(w2[i]);
  for (int i = tid; i < cpad * 128; i += stride) W3[i] = f2bf((i >> 7) < cout ? w3[i] : 0.f);
  for (int i = tid; i < cpad; i += stride) B3[i] = (i < cout) ? b3[i] : 0.f;
}

// ---------------- fused 3-layer MLP step (one (b, t-tile) per workgroup) ----
struct NetSel { const float* b1[4]; const float* b2[4]; int net[4]; };

__global__ __launch_bounds__(NTHR)
void mlp_step_kernel(const float* __restrict__ offp, const float* __restrict__ spp,
                     float* __restrict__ dec, const unsigned char* __restrict__ ws,
                     int step, NetSel sel) {
  const int net = sel.net[blockIdx.y];
  const float* b1 = sel.b1[blockIdx.y];
  const float* b2 = sel.b2[blockIdx.y];
  const int cin = c_cin[net], kpad = c_kpad[net], kst = c_kst[net];
  const int cout = c_cout[net][step], cpad = c_cpad[net][step];
  size_t o1, o2, o3, ob; ws_offsets(net, step, o1, o2, o3, ob);
  const unsigned short* W1 = (const unsigned short*)(ws + o1);
  const unsigned short* W2 = (const unsigned short*)(ws + o2);
  const unsigned short* W3 = (const unsigned short*)(ws + o3);
  const float* B3 = (const float*)(ws + ob);

  const int tile = blockIdx.x;
  const int b  = tile >> 5;             // 32 tiles of TM=128 per batch
  const int t0 = (tile & 31) * TM;

  extern __shared__ unsigned char lds[];
  int xbytes = TM * kst * 2;
  if (xbytes < TM * HSTRIDE * 2) xbytes = TM * HSTRIDE * 2;   // region reused for H2
  unsigned short* Xs = (unsigned short*)lds;                   // [t][k], stride kst
  unsigned short* H1 = (unsigned short*)(lds + xbytes);        // [t][h], stride HSTRIDE
  unsigned short* H2 = Xs;                                     // reuse, stride HSTRIDE
  float*          Y  = (float*)(lds + xbytes);                 // reuse H1 region, [t][o]

  const int tid = threadIdx.x;
  const float* bases[3] = {
    offp + (size_t)b * OFFC * TDIM + t0,
    dec  + (size_t)b * DECC * TDIM + t0,
    spp  + (size_t)b * SPC  * TDIM + t0
  };

  // ---- stage gathered input X into LDS as bf16 [t][k] (coalesced along t) ---
  for (int i = tid; i < TM * cin; i += NTHR) {
    int k = i >> 7, t = i & (TM - 1);
    int rem = k, s = 0;
    while (rem >= (int)c_segs[net][s].cnt) { rem -= c_segs[net][s].cnt; ++s; }
    const Seg sg = c_segs[net][s];
    float v = bases[sg.src][(size_t)(sg.start + rem) * TDIM + t];
    Xs[t * kst + k] = f2bf(v);
  }
  for (int i = TM * cin + tid; i < TM * kpad; i += NTHR) {     // K zero-pad
    int k = i >> 7, t = i & (TM - 1);
    Xs[t * kst + k] = 0;
  }
  __syncthreads();

  const int wave = tid >> 5, lane = tid & 31;
  const int l16 = lane & 15, hi = lane >> 4;
  const int arow = wave * 16 + l16;

  // A-matrix 16x32 bf16 fragment: lanes 0-15 K {kk..kk+7, kk+16..kk+23}, hi lanes +8
  auto loadA = [&](const unsigned short* base, int st, int kk) {
    const unsigned short* p = base + (size_t)arow * st + kk + hi * 8;
    union { v16bf v; v4f f[2]; } u;
    u.f[0] = *(const v4f*)p;
    u.f[1] = *(const v4f*)(p + 16);
    return u.v;
  };
  // B-matrix 32x16 bf16 fragment: lane = column n, K {kk..kk+7, kk+16..kk+23}, hi lanes +8
  auto loadB = [&](const unsigned short* base, int st, int nn, int kk) {
    const unsigned short* p = base + (size_t)(nn + l16) * st + kk + hi * 8;
    union { v16bf v; v4f f[2]; } u;
    u.f[0] = *(const v4f*)p;
    u.f[1] = *(const v4f*)(p + 16);
    return u.v;
  };

  // store one 16x16 f32 D tile to LDS as relu'd bf16 [t][h]
  auto storeH = [&](unsigned short* H, v8f acc, int col) {
#pragma unroll
    for (int v = 0; v < 8; ++v) {
      float x = acc[v];
      H[(wave * 16 + hi * 8 + v) * HSTRIDE + col] = f2bf(x > 0.f ? x : 0.f);
    }
  };

  // ---------------- layer 1: X(TMxcin) @ W1^T -> H1 (relu, bf16) ------------
  // 2-way N blocking: one A fragment feeds two independent WMMA chains
  for (int nt = 0; nt < 8; nt += 2) {
    float bv0 = b1[nt * 16 + l16];
    float bv1 = b1[nt * 16 + 16 + l16];
    v8f acc0 = {bv0, bv0, bv0, bv0, bv0, bv0, bv0, bv0};
    v8f acc1 = {bv1, bv1, bv1, bv1, bv1, bv1, bv1, bv1};
    for (int kk = 0; kk < kpad; kk += 32) {
      v16bf a = loadA(Xs, kst, kk);
      acc0 = __builtin_amdgcn_wmma_f32_16x16x32_bf16(false, a, false,
                 loadB(W1, kpad, nt * 16,      kk), (short)0, acc0, false, false);
      acc1 = __builtin_amdgcn_wmma_f32_16x16x32_bf16(false, a, false,
                 loadB(W1, kpad, nt * 16 + 16, kk), (short)0, acc1, false, false);
    }
    storeH(H1, acc0, nt * 16 + l16);
    storeH(H1, acc1, nt * 16 + 16 + l16);
  }
  __syncthreads();

  // ---------------- layer 2: H1 @ W2^T -> H2 (relu, bf16, reuses X region) --
  for (int nt = 0; nt < 8; nt += 2) {
    float bv0 = b2[nt * 16 + l16];
    float bv1 = b2[nt * 16 + 16 + l16];
    v8f acc0 = {bv0, bv0, bv0, bv0, bv0, bv0, bv0, bv0};
    v8f acc1 = {bv1, bv1, bv1, bv1, bv1, bv1, bv1, bv1};
#pragma unroll
    for (int kk = 0; kk < 128; kk += 32) {
      v16bf a = loadA(H1, HSTRIDE, kk);
      acc0 = __builtin_amdgcn_wmma_f32_16x16x32_bf16(false, a, false,
                 loadB(W2, 128, nt * 16,      kk), (short)0, acc0, false, false);
      acc1 = __builtin_amdgcn_wmma_f32_16x16x32_bf16(false, a, false,
                 loadB(W2, 128, nt * 16 + 16, kk), (short)0, acc1, false, false);
    }
    storeH(H2, acc0, nt * 16 + l16);
    storeH(H2, acc1, nt * 16 + 16 + l16);
  }
  __syncthreads();

  // ---------------- layer 3: H2 @ W3^T -> Y (f32, no relu) ------------------
  if (cpad == 32) {
    float bv0 = B3[l16];
    float bv1 = B3[16 + l16];
    v8f acc0 = {bv0, bv0, bv0, bv0, bv0, bv0, bv0, bv0};
    v8f acc1 = {bv1, bv1, bv1, bv1, bv1, bv1, bv1, bv1};
#pragma unroll
    for (int kk = 0; kk < 128; kk += 32) {
      v16bf a = loadA(H2, HSTRIDE, kk);
      acc0 = __builtin_amdgcn_wmma_f32_16x16x32_bf16(false, a, false,
                 loadB(W3, 128, 0,  kk), (short)0, acc0, false, false);
      acc1 = __builtin_amdgcn_wmma_f32_16x16x32_bf16(false, a, false,
                 loadB(W3, 128, 16, kk), (short)0, acc1, false, false);
    }
#pragma unroll
    for (int v = 0; v < 8; ++v) {
      Y[(wave * 16 + hi * 8 + v) * YSTRIDE + l16]      = acc0[v];
      Y[(wave * 16 + hi * 8 + v) * YSTRIDE + 16 + l16] = acc1[v];
    }
  } else {
    float bv0 = B3[l16];
    v8f acc0 = {bv0, bv0, bv0, bv0, bv0, bv0, bv0, bv0};
#pragma unroll
    for (int kk = 0; kk < 128; kk += 32) {
      v16bf a = loadA(H2, HSTRIDE, kk);
      acc0 = __builtin_amdgcn_wmma_f32_16x16x32_bf16(false, a, false,
                 loadB(W3, 128, 0, kk), (short)0, acc0, false, false);
    }
#pragma unroll
    for (int v = 0; v < 8; ++v)
      Y[(wave * 16 + hi * 8 + v) * YSTRIDE + l16] = acc0[v];
  }
  __syncthreads();

  // ---------------- dq-normalize groups of 8 + scatter into dec -------------
  for (int i = tid; i < cout * TM; i += NTHR) {
    int o = i >> 7, t = i & (TM - 1);
    const float* yr = Y + t * YSTRIDE;
    int g = o & ~7;
    float ss = yr[g] * yr[g] + yr[g + 1] * yr[g + 1] + yr[g + 2] * yr[g + 2] + yr[g + 3] * yr[g + 3];
    float val = yr[o] / sqrtf(ss);
    OSeg s0 = c_oseg[net][step][0];
    int ch = (o < s0.cnt) ? (s0.start + o) : (c_oseg[net][step][1].start + (o - s0.cnt));
    dec[(size_t)b * DECC * TDIM + (size_t)ch * TDIM + t0 + t] = val;
  }
}

// --------------------------------- host -------------------------------------
extern "C" void kernel_launch(void* const* d_in, const int* in_sizes, int n_in,
                              void* d_out, int out_size, void* d_ws, size_t ws_size,
                              hipStream_t stream) {
  const float* dec_in = nullptr; const float* sp_in = nullptr; const float* off_in = nullptr;
  int pidx[64]; int np = 0;
  for (int i = 0; i < n_in; ++i) {
    long long s = in_sizes[i];
    if      (s == (long long)NB * DECC * TDIM) dec_in = (const float*)d_in[i];
    else if (s == (long long)NB * SPC  * TDIM) sp_in  = (const float*)d_in[i];
    else if (s == (long long)NB * OFFC * TDIM) off_in = (const float*)d_in[i];
    else if (np < 64) pidx[np++] = i;
  }
  if (!dec_in || !sp_in || !off_in || np < 30) return;
  if (ws_size < 996672) return;   // packed-weight image must fit

  // Param leaves: canonical slot j = {w1,b1,w2,b2,w3,b3}, net = {main,lleg,rleg,larm,rarm}
  const float* Wt[5][3][6];
  if (in_sizes[pidx[0]] == 128 * 207) {
    // dict-insertion order: main..rarm, step 0..2, w1,b1,w2,b2,w3,b3
    int c = 0;
    for (int n = 0; n < 5; ++n)
      for (int s = 0; s < 3; ++s)
        for (int j = 0; j < 6; ++j) Wt[n][s][j] = (const float*)d_in[pidx[c++]];
  } else {
    // jax-sorted order: nets larm,lleg,main,rarm,rleg; leaves b1,b2,b3,w1,w2,w3
    static const int netOrder[5] = {3, 1, 0, 4, 2};
    static const int leafMap[6]  = {1, 3, 5, 0, 2, 4};
    int c = 0;
    for (int n = 0; n < 5; ++n)
      for (int s = 0; s < 3; ++s)
        for (int l = 0; l < 6; ++l) Wt[netOrder[n]][s][leafMap[l]] = (const float*)d_in[pidx[c++]];
  }

  // working copy of dec (never mutate inputs)
  hipMemcpyAsync(d_out, dec_in, (size_t)NB * DECC * TDIM * sizeof(float),
                 hipMemcpyDeviceToDevice, stream);

  unsigned char* ws = (unsigned char*)d_ws;
  for (int n = 0; n < 5; ++n)
    for (int s = 0; s < 3; ++s)
      pack_weights_kernel<<<dim3(112), dim3(NTHR), 0, stream>>>(
          Wt[n][s][0], Wt[n][s][2], Wt[n][s][4], Wt[n][s][5], ws, n, s);

  const int ldsMain = 57856 + 33280;   // X(128x226 bf16) + H1 region
  const int ldsLimb = 33280 + 33280;
  (void)hipFuncSetAttribute(reinterpret_cast<const void*>(mlp_step_kernel),
                            hipFuncAttributeMaxDynamicSharedMemorySize, ldsMain);

  // main: 3 dependent steps
  for (int s = 0; s < 3; ++s) {
    NetSel sel{};
    for (int j = 0; j < 4; ++j) { sel.net[j] = 0; sel.b1[j] = Wt[0][s][1]; sel.b2[j] = Wt[0][s][3]; }
    mlp_step_kernel<<<dim3(NB * (TDIM / TM), 1), dim3(NTHR), ldsMain, stream>>>(
        off_in, sp_in, (float*)d_out, ws, s, sel);
  }
  // limbs: 4 independent chains per round -> gridDim.y = 4
  for (int s = 0; s < 3; ++s) {
    NetSel sel{};
    for (int j = 0; j < 4; ++j) { sel.net[j] = j + 1; sel.b1[j] = Wt[j + 1][s][1]; sel.b2[j] = Wt[j + 1][s][3]; }
    mlp_step_kernel<<<dim3(NB * (TDIM / TM), 4), dim3(NTHR), ldsLimb, stream>>>(
        off_in, sp_in, (float*)d_out, ws, s, sel);
  }
}
